// SelfAttention_6700148981816
// MI455X (gfx1250) — matrix operand order repaired and checked by reading, not compiled, not run on hardware
//
#include <hip/hip_runtime.h>
#include <hip/hip_bf16.h>

#define KDIM 128
#define HEADS 8
#define BB 4
#define TT 2048
#define NBLK (TT / 32)  // 64 key blocks of 32

typedef __attribute__((ext_vector_type(16))) _Float16 v16h;
typedef __attribute__((ext_vector_type(8)))  _Float16 v8h;
typedef __attribute__((ext_vector_type(8)))  float    v8f;
typedef __attribute__((ext_vector_type(4)))  int      v4i;

// D = A(16x32 f16) * B(32x16 f16) + C(16x16 f32)
__device__ __forceinline__ v8f wmma16(v16h a, v16h b, v8f c) {
  return __builtin_amdgcn_wmma_f32_16x16x32_f16(false, a, false, b, (short)0, c,
                                                false, false);
}

// A-fragment (16x32 f16) from row-major tile, row stride lda (halves).
// lane m holds row m; g=0 lanes: K=0..15 ; g=1 lanes: K=16..31
__device__ __forceinline__ v16h ldA(const _Float16* tile, int lda, int lane) {
  const int m = lane & 15, g = lane >> 4;
  const _Float16* p = tile + (size_t)m * lda + g * 16;
  v8h lo = *(const v8h*)(p);       // K = g*16 + 0..7
  v8h hi = *(const v8h*)(p + 8);   // K = g*16 + 8..15
  v16h r;
#pragma unroll
  for (int i = 0; i < 8; ++i) { r[i] = lo[i]; r[i + 8] = hi[i]; }
  return r;
}

// B-fragment (32x16 f16): lane holds its column's 16 contraction values at p.
__device__ __forceinline__ v16h ldB(const _Float16* p) {
  v8h lo = *(const v8h*)(p);
  v8h hi = *(const v8h*)(p + 8);
  v16h r;
#pragma unroll
  for (int i = 0; i < 8; ++i) { r[i] = lo[i]; r[i + 8] = hi[i]; }
  return r;
}

// Async global->LDS 16-byte copy per lane (ASYNCcnt path), with sync fallback.
__device__ __forceinline__ void async_ld16(const _Float16* g, _Float16* l) {
#if __has_builtin(__builtin_amdgcn_global_load_async_to_lds_b128)
  __builtin_amdgcn_global_load_async_to_lds_b128(
      (__attribute__((address_space(1))) v4i*)g,
      (__attribute__((address_space(3))) v4i*)l, 0, 0);
#else
  *(v8h*)l = *(const v8h*)g;
#endif
}

template <int N>
__device__ __forceinline__ void wait_async() {
#if __has_builtin(__builtin_amdgcn_s_wait_asynccnt)
  __builtin_amdgcn_s_wait_asynccnt(N);
#else
  asm volatile("s_wait_asynccnt %0" ::"i"(N) : "memory");
#endif
}

__global__ void cvt_f32_f16(const float* __restrict__ src, _Float16* __restrict__ dst,
                            int n, float scale) {
  int i = blockIdx.x * blockDim.x + threadIdx.x;
  if (i < n) dst[i] = (_Float16)(src[i] * scale);
}

// q/k/v projection: one wave per 16x16 tile of one head's output.
// grid (256, B*H, 3), block (32,4). which: 0=q, 1=k, 2=v(transposed store).
__global__ __launch_bounds__(128) void qkv_proj(
    const _Float16* __restrict__ xh, const _Float16* __restrict__ Wqh,
    const _Float16* __restrict__ Wkh, const _Float16* __restrict__ Wvh,
    _Float16* __restrict__ qo, _Float16* __restrict__ ko,
    _Float16* __restrict__ vTo) {
  const int lane = threadIdx.x;
  const int n = lane & 15, g = lane >> 4;
  const int tile = blockIdx.x * 4 + threadIdx.y;  // 0..1023
  const int td = tile & 7, tt = tile >> 3;        // d-tile, t-tile
  const int bh = blockIdx.y;
  const int b = bh >> 3, h = bh & 7;
  const int which = blockIdx.z;
  const _Float16* W = (which == 0) ? Wqh : ((which == 1) ? Wkh : Wvh);

  const _Float16* abase = xh + ((size_t)(b * TT + tt * 16)) * KDIM;
  const _Float16* wrow = W + (size_t)(h * KDIM + td * 16 + n) * KDIM + g * 16;

  v8f acc = (v8f)0.0f;
#pragma unroll
  for (int kc = 0; kc < 4; ++kc) {
    v16h a = ldA(abase + kc * 32, KDIM, lane);
    v16h bm = ldB(wrow + kc * 32);
    acc = wmma16(a, bm, acc);
  }
  if (which < 2) {
    _Float16* out = (which == 0) ? qo : ko;  // [bh][t][d] row-major
    size_t base = ((size_t)bh * TT + tt * 16 + g * 8) * KDIM + td * 16 + n;
#pragma unroll
    for (int r = 0; r < 8; ++r) out[base + (size_t)r * KDIM] = (_Float16)acc[r];
  } else {  // V stored transposed: [bh][d][t]
    size_t base = ((size_t)bh * KDIM + td * 16 + n) * TT + tt * 16 + g * 8;
#pragma unroll
    for (int r = 0; r < 8; ++r) vTo[base + r] = (_Float16)acc[r];
  }
}

// Flash attention with double-buffered async global->LDS staging of K/V.
// Each wave owns 16 query rows; 4 waves share each staged 32-key block.
// grid (T/64, B*H), block (32,4).
__global__ __launch_bounds__(128) void flash_attn(
    const _Float16* __restrict__ qh, const _Float16* __restrict__ kh,
    const _Float16* __restrict__ vT, _Float16* __restrict__ oh) {
  __shared__ _Float16 Kt[2][32 * 128];    // [buf][s_local][d], 2 x 8KB
  __shared__ _Float16 Vl[2][128 * 32];    // [buf][d][s_local], 2 x 8KB
  __shared__ _Float16 plds[4 * 16 * 32];  // per-wave P staging, 4KB

  const int lane = threadIdx.x;
  const int n = lane & 15, g = lane >> 4;
  const int tid = threadIdx.y * 32 + lane;  // 0..127
  const int bh = blockIdx.y;
  const int b = bh >> 3, h = bh & 7;
  const int t0 = blockIdx.x * 64 + threadIdx.y * 16;
  _Float16* pl = plds + threadIdx.y * (16 * 32);

  const _Float16* kbh = kh + (size_t)bh * TT * KDIM;
  const _Float16* vbh = vT + (size_t)bh * KDIM * TT;

  // Stage one 32-key block (K: 8KB contiguous, V: 128 rows x 64B from vT).
  // Exactly 8 async instructions per wave per call.
  auto stage = [&](int buf, int sb) {
#pragma unroll
    for (int p = 0; p < 4; ++p) {
      int off = (p * 128 + tid) * 8;  // halves, 16B chunks
      async_ld16(kbh + (size_t)sb * KDIM + off, &Kt[buf][off]);
    }
#pragma unroll
    for (int p = 0; p < 4; ++p) {
      int c = p * 128 + tid;  // 0..511
      int d = c >> 2, cs = (c & 3) * 8;
      async_ld16(vbh + (size_t)d * TT + sb + cs, &Vl[buf][d * 32 + cs]);
    }
  };

  // Q fragments (16 rows x 128 d) held in registers for the whole loop.
  const _Float16* qbase = qh + ((size_t)bh * TT + t0) * KDIM;
  v16h qf[4];
#pragma unroll
  for (int kc = 0; kc < 4; ++kc) qf[kc] = ldA(qbase + kc * 32, KDIM, lane);

  v8f o[8];
#pragma unroll
  for (int i = 0; i < 8; ++i) o[i] = (v8f)0.0f;
  float mrow[8], lpart[8];
#pragma unroll
  for (int r = 0; r < 8; ++r) { mrow[r] = -1e30f; lpart[r] = 0.0f; }

  // Prologue: blocks 0 and 1 in flight; wait for block 0 (in-order completion).
  stage(0, 0);
  stage(1, 32);
  wait_async<8>();
  __syncthreads();

  for (int i = 0; i < NBLK; ++i) {
    const int cur = i & 1;
    const _Float16* kb = Kt[cur];
    const _Float16* vb = Vl[cur];

    // S = Q K^T for 32 staged key columns (two 16x16 f32 tiles)
    v8f S0 = (v8f)0.0f, S1 = (v8f)0.0f;
#pragma unroll
    for (int kc = 0; kc < 4; ++kc) {
      v16h b0 = ldB(kb + (size_t)n * KDIM + kc * 32 + g * 16);
      v16h b1 = ldB(kb + (size_t)(n + 16) * KDIM + kc * 32 + g * 16);
      S0 = wmma16(qf[kc], b0, S0);
      S1 = wmma16(qf[kc], b1, S1);
    }

    // Online softmax. C-layout: lanes 0-15 rows 0-7, lanes 16-31 rows 8-15.
    float corr[8];
#pragma unroll
    for (int r = 0; r < 8; ++r) {
      float t = fmaxf(S0[r], S1[r]);
      t = fmaxf(t, __shfl_xor(t, 1));
      t = fmaxf(t, __shfl_xor(t, 2));
      t = fmaxf(t, __shfl_xor(t, 4));
      t = fmaxf(t, __shfl_xor(t, 8));
      float mn = fmaxf(mrow[r], t);
      corr[r] = __expf(mrow[r] - mn);
      mrow[r] = mn;
      float p0 = __expf(S0[r] - mn);
      float p1 = __expf(S1[r] - mn);
      lpart[r] = lpart[r] * corr[r] + p0 + p1;
      pl[(g * 8 + r) * 32 + n] = (_Float16)p0;
      pl[(g * 8 + r) * 32 + 16 + n] = (_Float16)p1;
    }
#pragma unroll
    for (int dt = 0; dt < 8; ++dt)
#pragma unroll
      for (int r = 0; r < 8; ++r) o[dt][r] *= corr[r];

    // O += P V from staged LDS tiles
    v16h pf = ldA(pl, 32, lane);
#pragma unroll
    for (int dt = 0; dt < 8; ++dt) {
      v16h vfrag = ldB(vb + (dt * 16 + n) * 32 + g * 16);
      o[dt] = wmma16(pf, vfrag, o[dt]);
    }

    __syncthreads();  // all waves done reading buf `cur`
    if (i + 2 < NBLK) {
      stage(cur, (i + 2) * 32);  // overlap DMA of block i+2 with next compute
      wait_async<8>();           // block i+1 resident (in-order)
    } else {
      wait_async<0>();           // drain tail
    }
    __syncthreads();  // block i+1 visible to all waves
  }

  // Normalize by row sums and store as f16 [b][t][h*128+d]
  float inv[8];
#pragma unroll
  for (int r = 0; r < 8; ++r) {
    float l = lpart[r];
    l += __shfl_xor(l, 1);
    l += __shfl_xor(l, 2);
    l += __shfl_xor(l, 4);
    l += __shfl_xor(l, 8);
    inv[r] = 1.0f / l;
  }
  const int HK = HEADS * KDIM;
  size_t obase = ((size_t)b * TT + t0 + g * 8) * HK + h * KDIM + n;
#pragma unroll
  for (int dt = 0; dt < 8; ++dt)
#pragma unroll
    for (int r = 0; r < 8; ++r)
      oh[obase + (size_t)r * HK + dt * 16] = (_Float16)(o[dt][r] * inv[r]);
}

// out = oh(8192x1024 f16) @ Wu^T + bu, f32 result. grid (1024), block (32,4).
__global__ __launch_bounds__(128) void out_proj(
    const _Float16* __restrict__ oh, const _Float16* __restrict__ Wuh,
    const float* __restrict__ bu, float* __restrict__ out) {
  const int lane = threadIdx.x;
  const int n = lane & 15, g = lane >> 4;
  const int tile = blockIdx.x * 4 + threadIdx.y;  // 0..4095
  const int jt = tile & 7, mt = tile >> 3;
  const int HK = HEADS * KDIM;

  const _Float16* abase = oh + (size_t)mt * 16 * HK;
  const _Float16* wrow = Wuh + (size_t)(jt * 16 + n) * HK + g * 16;

  v8f acc = (v8f)0.0f;
  for (int kc = 0; kc < 32; ++kc) {
    v16h a = ldA(abase + kc * 32, HK, lane);
    v16h bm = ldB(wrow + kc * 32);
    acc = wmma16(a, bm, acc);
  }
  float bias = bu[jt * 16 + n];
  size_t base = (size_t)(mt * 16 + g * 8) * KDIM + jt * 16 + n;
#pragma unroll
  for (int r = 0; r < 8; ++r) out[base + (size_t)r * KDIM] = acc[r] + bias;
}

extern "C" void kernel_launch(void* const* d_in, const int* in_sizes, int n_in,
                              void* d_out, int out_size, void* d_ws, size_t ws_size,
                              hipStream_t stream) {
  const float* x  = (const float*)d_in[0];
  const float* Wq = (const float*)d_in[1];
  const float* Wk = (const float*)d_in[2];
  const float* Wv = (const float*)d_in[3];
  const float* Wu = (const float*)d_in[4];
  const float* bu = (const float*)d_in[5];
  float* out = (float*)d_out;

  const int NX = BB * TT * KDIM;          // 1,048,576
  const int NW = HEADS * KDIM * KDIM;     // 131,072 (also Wu size)
  const int NQ = BB * HEADS * TT * KDIM;  // 8,388,608

  // Workspace layout (f16 halves), total ~67 MB
  _Float16* xh  = (_Float16*)d_ws;
  _Float16* Wqh = xh + NX;
  _Float16* Wkh = Wqh + NW;
  _Float16* Wvh = Wkh + NW;
  _Float16* Wuh = Wvh + NW;
  _Float16* qh  = Wuh + NW;
  _Float16* kh  = qh + NQ;
  _Float16* vT  = kh + NQ;
  _Float16* oh  = vT + NQ;

  const float qkScale = 0.29730177875068026f;  // 128^(-1/4), folded into Wq,Wk

  cvt_f32_f16<<<(NX + 255) / 256, 256, 0, stream>>>(x, xh, NX, 1.0f);
  cvt_f32_f16<<<(NW + 255) / 256, 256, 0, stream>>>(Wq, Wqh, NW, qkScale);
  cvt_f32_f16<<<(NW + 255) / 256, 256, 0, stream>>>(Wk, Wkh, NW, qkScale);
  cvt_f32_f16<<<(NW + 255) / 256, 256, 0, stream>>>(Wv, Wvh, NW, 1.0f);
  cvt_f32_f16<<<(NW + 255) / 256, 256, 0, stream>>>(Wu, Wuh, NW, 1.0f);

  qkv_proj<<<dim3(256, BB * HEADS, 3), dim3(32, 4), 0, stream>>>(
      xh, Wqh, Wkh, Wvh, qh, kh, vT);
  flash_attn<<<dim3(TT / 64, BB * HEADS), dim3(32, 4), 0, stream>>>(qh, kh, vT, oh);
  out_proj<<<dim3(1024), dim3(32, 4), 0, stream>>>(oh, Wuh, bu, out);
}
